// GATDepthNet_7713761263902
// MI455X (gfx1250) — compile-verified, matrix operand order checked by reading, not run on hardware
//
#include <hip/hip_runtime.h>
#include <hip/hip_bf16.h>

#define HEADS 8
#define CH 16
#define EMBED 128
#define NEG_SLOPE 0.2f

typedef __bf16 v8bf  __attribute__((ext_vector_type(8)));
typedef __bf16 v16bf __attribute__((ext_vector_type(16)));
typedef float  v8f   __attribute__((ext_vector_type(8)));

union V16U { v16bf v; v8bf h[2]; };

// ---------- helpers ----------
__device__ __forceinline__ unsigned short f2bf(float f) {
    unsigned u = __float_as_uint(f);
    unsigned r = u + 0x7FFFu + ((u >> 16) & 1u);   // round-to-nearest-even
    return (unsigned short)(r >> 16);
}
__device__ __forceinline__ unsigned enc_ord(float x) {   // monotonic uint encoding
    unsigned u = __float_as_uint(x);
    return (u & 0x80000000u) ? ~u : (u | 0x80000000u);
}
__device__ __forceinline__ float dec_ord(unsigned u) {
    return (u & 0x80000000u) ? __uint_as_float(u & 0x7FFFFFFFu)
                             : __uint_as_float(~u);
}
__device__ __forceinline__ float lrelu(float v) { return v > 0.f ? v : NEG_SLOPE * v; }

// ---------- conversions ----------
__global__ void cvt_x_bf16(const float* __restrict__ x, unsigned short* __restrict__ xb, int n) {
    int i = blockIdx.x * blockDim.x + threadIdx.x;
    if (i < n) xb[i] = f2bf(x[i]);
}
// W[k][n] (in x out, row-major) -> Wt_bf[n][k]
__global__ void cvt_w_bf16(const float* __restrict__ W, unsigned short* __restrict__ Wt) {
    int i = blockIdx.x * blockDim.x + threadIdx.x;          // i = n*128 + k
    if (i < EMBED * EMBED) {
        int n = i >> 7, k = i & 127;
        Wt[i] = f2bf(W[k * EMBED + n]);
    }
}

// ---------- per-layer init: m=ord(-inf), denom=0, out=0 ----------
__global__ void init_layer(unsigned* __restrict__ mbits, float* __restrict__ denom,
                           float* __restrict__ out, int N) {
    int i = blockIdx.x * blockDim.x + threadIdx.x;
    if (i < N * HEADS) { mbits[i] = 0u; denom[i] = 0.f; }
    if (i < N * EMBED) out[i] = 0.f;
}

// ---------- WMMA GEMM: H[N,128] = Xbf[N,128] @ W (via Wt[n][k]) ----------
__global__ void gemm_bf16_wmma(const unsigned short* __restrict__ Abf,
                               const unsigned short* __restrict__ Btbf,
                               float* __restrict__ H, int nrows) {
    const int wave = threadIdx.x >> 5;          // 0..7  -> output column tile
    const int lane = threadIdx.x & 31;
    const int row0 = blockIdx.x * 16;
    const int m    = lane & 15;
    const int ksel = (lane >> 4) << 3;          // 0 or 8 (bf16 A/B lane striping)

    const unsigned short* arow = Abf  + (size_t)(row0 + m) * EMBED;
    const unsigned short* brow = Btbf + (size_t)(wave * 16 + m) * EMBED;

    v8f c = {};
#pragma unroll
    for (int kb = 0; kb < EMBED; kb += 32) {
        V16U a, b;
        a.h[0] = *reinterpret_cast<const v8bf*>(arow + kb + ksel);
        a.h[1] = *reinterpret_cast<const v8bf*>(arow + kb + ksel + 16);
        b.h[0] = *reinterpret_cast<const v8bf*>(brow + kb + ksel);
        b.h[1] = *reinterpret_cast<const v8bf*>(brow + kb + ksel + 16);
        c = __builtin_amdgcn_wmma_f32_16x16x32_bf16(false, a.v, false, b.v,
                                                    (short)0, c, false, false);
    }
    // C/D layout: VGPR g -> M = g + (lane<16 ? 0 : 8), N = lane&15
    const int n     = wave * 16 + (lane & 15);
    const int mbase = (lane >> 4) << 3;
    float* hp = H + (size_t)(row0 + mbase) * EMBED + n;
    if (row0 + 16 <= nrows) {
        // full block (always taken when nrows % 16 == 0): branch-free stores
#pragma unroll
        for (int g = 0; g < 8; ++g)
            hp[(size_t)g * EMBED] = c[g];
    } else {
        // ragged tail block (not hit for N=100000, kept for generality)
#pragma unroll
        for (int g = 0; g < 8; ++g)
            if (row0 + mbase + g < nrows) hp[(size_t)g * EMBED] = c[g];
    }
}

// ---------- per-node attention logits ----------
__global__ void attn_logits(const float* __restrict__ H,
                            const float* __restrict__ a_s, const float* __restrict__ a_d,
                            float* __restrict__ als, float* __restrict__ ald, int N) {
    int i = blockIdx.x * blockDim.x + threadIdx.x;   // node*HEADS + h
    if (i >= N * HEADS) return;
    int hd = i & (HEADS - 1);
    int node = i >> 3;
    const float* hp = H + (size_t)node * EMBED + hd * CH;
    float ss = 0.f, sd = 0.f;
#pragma unroll
    for (int c = 0; c < CH; ++c) {
        float v = hp[c];
        ss += v * a_s[hd * CH + c];
        sd += v * a_d[hd * CH + c];
    }
    als[i] = ss;
    ald[i] = sd;
}

// ---------- edge pass 1: segment max (ordered-uint atomics) ----------
__global__ void edge_max(const int* __restrict__ ei, int E, int N,
                         const float* __restrict__ als, const float* __restrict__ ald,
                         unsigned* __restrict__ mbits) {
    int e = blockIdx.x * blockDim.x + threadIdx.x;
    if (e >= E + N) return;
    int s, d;
    if (e < E) { s = ei[e]; d = ei[E + e]; } else { s = d = e - E; }
    const float4* ps = (const float4*)(als + (size_t)s * HEADS);
    const float4* pd = (const float4*)(ald + (size_t)d * HEADS);
    float4 a0 = ps[0], a1 = ps[1], b0 = pd[0], b1 = pd[1];
    float v[HEADS] = { a0.x + b0.x, a0.y + b0.y, a0.z + b0.z, a0.w + b0.w,
                       a1.x + b1.x, a1.y + b1.y, a1.z + b1.z, a1.w + b1.w };
    unsigned* mp = mbits + (size_t)d * HEADS;
#pragma unroll
    for (int h = 0; h < HEADS; ++h)
        atomicMax(&mp[h], enc_ord(lrelu(v[h])));
}

// ---------- edge pass 2: segment sum of exp ----------
__global__ void edge_denom(const int* __restrict__ ei, int E, int N,
                           const float* __restrict__ als, const float* __restrict__ ald,
                           const unsigned* __restrict__ mbits, float* __restrict__ denom) {
    int e = blockIdx.x * blockDim.x + threadIdx.x;
    if (e >= E + N) return;
    int s, d;
    if (e < E) { s = ei[e]; d = ei[E + e]; } else { s = d = e - E; }
    const float4* ps = (const float4*)(als + (size_t)s * HEADS);
    const float4* pd = (const float4*)(ald + (size_t)d * HEADS);
    float4 a0 = ps[0], a1 = ps[1], b0 = pd[0], b1 = pd[1];
    float v[HEADS] = { a0.x + b0.x, a0.y + b0.y, a0.z + b0.z, a0.w + b0.w,
                       a1.x + b1.x, a1.y + b1.y, a1.z + b1.z, a1.w + b1.w };
    const unsigned* mp = mbits + (size_t)d * HEADS;
    float* dp = denom + (size_t)d * HEADS;
#pragma unroll
    for (int h = 0; h < HEADS; ++h)
        atomicAdd(&dp[h], __expf(lrelu(v[h]) - dec_ord(mp[h])));
}

// ---------- edge pass 3: weighted scatter-add, one wave32 per edge ----------
__global__ void aggregate(const int* __restrict__ ei, int E, int N,
                          const float* __restrict__ als, const float* __restrict__ ald,
                          const unsigned* __restrict__ mbits, const float* __restrict__ denom,
                          const float* __restrict__ H, float* __restrict__ out) {
    int gid  = blockIdx.x * blockDim.x + threadIdx.x;
    int e    = gid >> 5;
    int lane = gid & 31;
    if (e >= E + N) return;
    int s, d;
    if (e < E) { s = ei[e]; d = ei[E + e]; } else { s = d = e - E; }
    int head = lane >> 2;                                   // 4 channels per lane
    float lv = lrelu(als[(size_t)s * HEADS + head] + ald[(size_t)d * HEADS + head]);
    float m  = dec_ord(mbits[(size_t)d * HEADS + head]);
    float alpha = __expf(lv - m) / (denom[(size_t)d * HEADS + head] + 1e-16f);
    float4 hv = *(const float4*)(H + (size_t)s * EMBED + lane * 4);
    float* op = out + (size_t)d * EMBED + lane * 4;
    atomicAdd(op + 0, hv.x * alpha);
    atomicAdd(op + 1, hv.y * alpha);
    atomicAdd(op + 2, hv.z * alpha);
    atomicAdd(op + 3, hv.w * alpha);
}

// ---------- bias + ELU (in place) ----------
__global__ void elu_bias(float* __restrict__ x, const float* __restrict__ b, int n) {
    int i = blockIdx.x * blockDim.x + threadIdx.x;
    if (i >= n) return;
    float v = x[i] + b[i & (EMBED - 1)];
    x[i] = v > 0.f ? v : (__expf(v) - 1.f);
}

// ---------- head: y[n] = x[n,:] . W + b, one wave per node ----------
__global__ void head_proj(const float* __restrict__ X, const float* __restrict__ Wh,
                          const float* __restrict__ bh, float* __restrict__ y, int N) {
    int gid  = blockIdx.x * blockDim.x + threadIdx.x;
    int node = gid >> 5;
    int lane = gid & 31;
    if (node >= N) return;
    float4 xv = *(const float4*)(X + (size_t)node * EMBED + lane * 4);
    float4 wv = *(const float4*)(Wh + lane * 4);
    float p = xv.x * wv.x + xv.y * wv.y + xv.z * wv.z + xv.w * wv.w;
#pragma unroll
    for (int off = 16; off > 0; off >>= 1) p += __shfl_down(p, off, 32);
    if (lane == 0) y[node] = p + bh[0];
}

// ---------- host launch ----------
extern "C" void kernel_launch(void* const* d_in, const int* in_sizes, int n_in,
                              void* d_out, int out_size, void* d_ws, size_t ws_size,
                              hipStream_t stream) {
    const float* x       = (const float*)d_in[0];
    const int*   ei      = (const int*)  d_in[1];
    const float* Ws      = (const float*)d_in[2];
    const float* att_src = (const float*)d_in[3];
    const float* att_dst = (const float*)d_in[4];
    const float* bs      = (const float*)d_in[5];
    const float* headW   = (const float*)d_in[6];
    const float* headb   = (const float*)d_in[7];

    const int N = in_sizes[0] / EMBED;
    const int E = in_sizes[1] / 2;
    const int DEPTH = 3;

    // carve workspace (256B aligned slices)
    auto align256 = [](size_t v) { return (v + 255) & ~(size_t)255; };
    char* p = (char*)d_ws;
    size_t szNE = align256((size_t)N * EMBED * sizeof(float));
    float* buf0 = (float*)p; p += szNE;
    float* buf1 = (float*)p; p += szNE;
    float* buf2 = (float*)p; p += szNE;
    unsigned short* xbf = (unsigned short*)p; p += align256((size_t)N * EMBED * sizeof(unsigned short));
    unsigned short* wbf = (unsigned short*)p; p += align256((size_t)EMBED * EMBED * sizeof(unsigned short));
    size_t szNH = align256((size_t)N * HEADS * sizeof(float));
    float*    als   = (float*)p;    p += szNH;
    float*    ald   = (float*)p;    p += szNH;
    unsigned* mbits = (unsigned*)p; p += szNH;
    float*    denom = (float*)p;    p += szNH;

    // layer input starts as a copy of x (don't mutate inputs)
    hipMemcpyAsync(buf0, x, (size_t)N * EMBED * sizeof(float),
                   hipMemcpyDeviceToDevice, stream);

    float* cur = buf0;   // layer input
    float* hb  = buf1;   // transformed features
    float* ob  = buf2;   // aggregation output

    const int T = 256;
    const int EN = E + N;
    for (int l = 0; l < DEPTH; ++l) {
        cvt_w_bf16<<<(EMBED * EMBED + T - 1) / T, T, 0, stream>>>(Ws + (size_t)l * EMBED * EMBED, wbf);
        cvt_x_bf16<<<(N * EMBED + T - 1) / T, T, 0, stream>>>(cur, xbf, N * EMBED);
        init_layer<<<(N * EMBED + T - 1) / T, T, 0, stream>>>(mbits, denom, ob, N);
        gemm_bf16_wmma<<<(N + 15) / 16, T, 0, stream>>>(xbf, wbf, hb, N);
        attn_logits<<<(N * HEADS + T - 1) / T, T, 0, stream>>>(
            hb, att_src + (size_t)l * EMBED, att_dst + (size_t)l * EMBED, als, ald, N);
        edge_max<<<(EN + T - 1) / T, T, 0, stream>>>(ei, E, N, als, ald, mbits);
        edge_denom<<<(EN + T - 1) / T, T, 0, stream>>>(ei, E, N, als, ald, mbits, denom);
        aggregate<<<((size_t)EN * 32 + T - 1) / T, T, 0, stream>>>(
            ei, E, N, als, ald, mbits, denom, hb, ob);
        elu_bias<<<(N * EMBED + T - 1) / T, T, 0, stream>>>(ob, bs + (size_t)l * EMBED, N * EMBED);
        // rotate: next input = ob, reuse old input as next output accumulator
        float* t = cur; cur = ob; ob = t;
    }

    head_proj<<<((size_t)N * 32 + T - 1) / T, T, 0, stream>>>(cur, headW, headb, (float*)d_out, N);
}
